// Decoder_11390253269443
// MI455X (gfx1250) — matrix operand order repaired and checked by reading, not compiled, not run on hardware
//
#include <hip/hip_runtime.h>
#include <math.h>

// Problem constants (match reference)
#define BATCH 128
#define LSEQ  256
#define EDIM  256
#define HDIM  256

typedef __attribute__((ext_vector_type(16))) _Float16 v16h;
typedef __attribute__((ext_vector_type(8)))  _Float16 v8h;
typedef __attribute__((ext_vector_type(8)))  float    v8f;

// ---------------------------------------------------------------------------
// WMMA GEMM, f16 inputs / f32 accumulate:
//   C[M,N] = A[M,K](f16) @ W[N,K](f16)^T + bias[N](f32)
// One wave computes a 1x4 strip of 16x16 tiles (A fragment reused 4x).
// All 10 loads of a k-step are issued before the 4 WMMAs so the scheduler
// can use partial s_wait_loadcnt thresholds and overlap load latency with
// matrix ops. 8 waves / 256 threads per block. K % 32 == 0, N % 64 == 0.
// ---------------------------------------------------------------------------
__global__ void gemm_wmma_f16(const _Float16* __restrict__ A,
                              const _Float16* __restrict__ W,
                              const float* __restrict__ bias,
                              float* __restrict__ C, int M, int N, int K) {
    const int wave = threadIdx.x >> 5;
    const int lane = threadIdx.x & 31;
    const int half = lane >> 4;
    const int rc   = lane & 15;

    const int groupsN = N >> 6;                   // 4 tiles per group
    const int task    = blockIdx.x * (blockDim.x >> 5) + wave;
    const int total   = (M >> 4) * groupsN;
    if (task >= total) return;                    // uniform per wave
    const int m0 = (task / groupsN) << 4;
    const int n0 = (task % groupsN) << 6;

    const _Float16* arow = A + (size_t)(m0 + rc) * K;
    const _Float16* wrow0 = W + (size_t)(n0 + rc) * K + (half << 4);
    const _Float16* wrow1 = wrow0 + (size_t)16 * K;
    const _Float16* wrow2 = wrow1 + (size_t)16 * K;
    const _Float16* wrow3 = wrow2 + (size_t)16 * K;

    v8f acc[4] = {};
    for (int k0 = 0; k0 < K; k0 += 32) {
        // Issue every load of this k-step before any WMMA consumes them.
        v8h alo = *(const v8h*)(arow + k0 + (half << 4));
        v8h ahi = *(const v8h*)(arow + k0 + 8 + (half << 4));
        v8h b0l = *(const v8h*)(wrow0 + k0);
        v8h b0h = *(const v8h*)(wrow0 + k0 + 8);
        v8h b1l = *(const v8h*)(wrow1 + k0);
        v8h b1h = *(const v8h*)(wrow1 + k0 + 8);
        v8h b2l = *(const v8h*)(wrow2 + k0);
        v8h b2h = *(const v8h*)(wrow2 + k0 + 8);
        v8h b3l = *(const v8h*)(wrow3 + k0);
        v8h b3h = *(const v8h*)(wrow3 + k0 + 8);

        v16h a = __builtin_shufflevector(alo, ahi, 0, 1, 2, 3, 4, 5, 6, 7,
                                         8, 9, 10, 11, 12, 13, 14, 15);
        v16h b;
        b = __builtin_shufflevector(b0l, b0h, 0, 1, 2, 3, 4, 5, 6, 7,
                                    8, 9, 10, 11, 12, 13, 14, 15);
        acc[0] = __builtin_amdgcn_wmma_f32_16x16x32_f16(
            false, a, false, b, (short)0, acc[0], false, false);
        b = __builtin_shufflevector(b1l, b1h, 0, 1, 2, 3, 4, 5, 6, 7,
                                    8, 9, 10, 11, 12, 13, 14, 15);
        acc[1] = __builtin_amdgcn_wmma_f32_16x16x32_f16(
            false, a, false, b, (short)0, acc[1], false, false);
        b = __builtin_shufflevector(b2l, b2h, 0, 1, 2, 3, 4, 5, 6, 7,
                                    8, 9, 10, 11, 12, 13, 14, 15);
        acc[2] = __builtin_amdgcn_wmma_f32_16x16x32_f16(
            false, a, false, b, (short)0, acc[2], false, false);
        b = __builtin_shufflevector(b3l, b3h, 0, 1, 2, 3, 4, 5, 6, 7,
                                    8, 9, 10, 11, 12, 13, 14, 15);
        acc[3] = __builtin_amdgcn_wmma_f32_16x16x32_f16(
            false, a, false, b, (short)0, acc[3], false, false);
    }

#pragma unroll
    for (int nt = 0; nt < 4; ++nt) {
        int ncol = n0 + (nt << 4) + rc;
        float bv = bias ? bias[ncol] : 0.0f;
#pragma unroll
        for (int r = 0; r < 8; ++r) {
            int m = m0 + r + (half << 3);          // C/D layout: reg r -> M=r(+8)
            C[(size_t)m * N + ncol] = acc[nt][r] + bv;
        }
    }
}

// ---------------------------------------------------------------------------
// One-time weight/context packing to f16
// ---------------------------------------------------------------------------
__global__ void pack_f16(const float* __restrict__ in, _Float16* __restrict__ out,
                         int n) {
    int i = blockIdx.x * blockDim.x + threadIdx.x;
    if (i < n) out[i] = (_Float16)in[i];
}

// Fuse [Wih | Whh] -> Wg16 [4H, 512] and bg = bih + bhh
__global__ void pack_gate_weights(const float* __restrict__ Wih,
                                  const float* __restrict__ Whh,
                                  const float* __restrict__ bih,
                                  const float* __restrict__ bhh,
                                  _Float16* __restrict__ Wg16,
                                  float* __restrict__ bg) {
    int i = blockIdx.x * blockDim.x + threadIdx.x;
    int total = 4 * HDIM * (EDIM + HDIM);
    if (i < total) {
        int n = i >> 9;                 // row (512 cols)
        int k = i & 511;
        float v = (k < EDIM) ? Wih[(size_t)n * EDIM + k]
                             : Whh[(size_t)n * HDIM + (k - EDIM)];
        Wg16[i] = (_Float16)v;
    }
    if (i < 4 * HDIM) bg[i] = bih[i] + bhh[i];
}

// ---------------------------------------------------------------------------
// LSTM elementwise: gates [B,4H] (i,f,g,o) -> c (in place), h_t (f32 + f16)
// ---------------------------------------------------------------------------
__device__ __forceinline__ float sigf(float v) { return 1.0f / (1.0f + expf(-v)); }

__global__ void lstm_elem(const float* __restrict__ gates,
                          float* __restrict__ c, float* __restrict__ h_t,
                          _Float16* __restrict__ ht16) {
    int idx = blockIdx.x * blockDim.x + threadIdx.x;
    if (idx >= BATCH * HDIM) return;
    int b = idx >> 8, hh = idx & (HDIM - 1);
    const float* g = gates + (size_t)b * 4 * HDIM;
    float ig = sigf(g[hh]);
    float fg = sigf(g[HDIM + hh]);
    float gg = tanhf(g[2 * HDIM + hh]);
    float og = sigf(g[3 * HDIM + hh]);
    float ct = fg * c[idx] + ig * gg;
    float ht = og * tanhf(ct);
    c[idx]    = ct;
    h_t[idx]  = ht;
    ht16[idx] = (_Float16)ht;
}

// ---------------------------------------------------------------------------
// Fused attention step (one block per batch row, 256 threads):
//   att = V . tanh(inp + ctx), mask, softmax -> alpha (written to outputs),
//   argmax pointer, mask update, input gather (-> xh16 f16),
//   hs = ctx^T alpha, h_new = tanh([hs|h_t] @ Who^T + bho) (-> hbuf + xh16)
// ---------------------------------------------------------------------------
__global__ void attn_step(const float* __restrict__ ctx, const float* __restrict__ inp,
                          const float* __restrict__ Vv, float* __restrict__ mask,
                          const float* __restrict__ h_t, const float* __restrict__ Who,
                          const float* __restrict__ bho, const float* __restrict__ emb,
                          float* __restrict__ h, _Float16* __restrict__ xh16,
                          float* __restrict__ out, int step) {
    __shared__ float s_inp[HDIM];
    __shared__ float s_att[LSEQ];
    __shared__ float s_red[LSEQ];
    __shared__ float s_hs[HDIM];
    __shared__ float s_ht[HDIM];
    __shared__ int   s_ridx[LSEQ];

    const int b = blockIdx.x;
    const int t = threadIdx.x;

    s_inp[t] = inp[b * HDIM + t];
    s_ht[t]  = h_t[b * HDIM + t];
    float mval = mask[b * LSEQ + t];
    __syncthreads();

    // attention energy for position l = t
    const float* cr = ctx + ((size_t)b * LSEQ + t) * HDIM;
    float acc = 0.0f;
    for (int k = 0; k < HDIM; ++k)
        acc += Vv[k] * tanhf(s_inp[k] + cr[k]);
    float att = (mval == 0.0f) ? -1e9f : acc;
    s_att[t] = att;
    s_red[t] = att;
    __syncthreads();

    // softmax: max reduction
    for (int s = LSEQ >> 1; s > 0; s >>= 1) {
        if (t < s) s_red[t] = fmaxf(s_red[t], s_red[t + s]);
        __syncthreads();
    }
    float mx = s_red[0];
    __syncthreads();
    float ev = expf(att - mx);
    s_red[t] = ev;
    __syncthreads();
    for (int s = LSEQ >> 1; s > 0; s >>= 1) {
        if (t < s) s_red[t] += s_red[t + s];
        __syncthreads();
    }
    float alpha = ev / s_red[0];
    __syncthreads();
    s_att[t] = alpha;
    // write alpha into outputs[b, step, t]
    out[((size_t)b * LSEQ + step) * LSEQ + t] = alpha;

    // pointer argmax over alpha * mask (strict > keeps lowest index on ties)
    s_red[t]  = alpha * mval;
    s_ridx[t] = t;
    __syncthreads();
    for (int s = LSEQ >> 1; s > 0; s >>= 1) {
        if (t < s) {
            if (s_red[t + s] > s_red[t]) {
                s_red[t]  = s_red[t + s];
                s_ridx[t] = s_ridx[t + s];
            }
        }
        __syncthreads();
    }
    int idx = s_ridx[0];

    // hs[h=t] = sum_l alpha[l] * ctx[b,l,t]
    const float* cb = ctx + (size_t)b * LSEQ * HDIM + t;
    float hsv = 0.0f;
    for (int l = 0; l < LSEQ; ++l)
        hsv += s_att[l] * cb[(size_t)l * HDIM];
    s_hs[t] = hsv;

    // mask update, next input gather (f16 into xh16 low half), pointer output
    if (t == 0) {
        mask[b * LSEQ + idx] = 0.0f;
        out[(size_t)BATCH * LSEQ * LSEQ + b * LSEQ + step] = (float)idx;
    }
    xh16[(size_t)b * (EDIM + HDIM) + t] =
        (_Float16)emb[((size_t)b * LSEQ + idx) * EDIM + t];
    __syncthreads();

    // h_new[t] = tanh([hs | h_t] @ Who^T + bho)
    const float* wr = Who + (size_t)t * (2 * HDIM);
    float a2 = bho[t];
    for (int j = 0; j < HDIM; ++j) a2 += wr[j] * s_hs[j];
    for (int j = 0; j < HDIM; ++j) a2 += wr[HDIM + j] * s_ht[j];
    float hn = tanhf(a2);
    h[b * HDIM + t] = hn;                                   // f32 (final output)
    xh16[(size_t)b * (EDIM + HDIM) + EDIM + t] = (_Float16)hn;  // f16 (gates GEMM)
}

// ---------------------------------------------------------------------------
// init / finalize
// ---------------------------------------------------------------------------
__global__ void init_state(float* __restrict__ h, float* __restrict__ c,
                           _Float16* __restrict__ xh16, float* __restrict__ mask,
                           const float* __restrict__ h0, const float* __restrict__ c0,
                           const float* __restrict__ dec) {
    int i = blockIdx.x * blockDim.x + threadIdx.x;
    if (i >= BATCH * HDIM) return;   // B*H == B*L == B*E == 32768
    int b = i >> 8, t = i & 255;
    h[i] = h0[i];
    c[i] = c0[i];
    mask[i] = 1.0f;
    xh16[(size_t)b * (EDIM + HDIM) + t]        = (_Float16)dec[i];
    xh16[(size_t)b * (EDIM + HDIM) + EDIM + t] = (_Float16)h0[i];
}

__global__ void finalize(const float* __restrict__ h, const float* __restrict__ c,
                         float* __restrict__ out) {
    int i = blockIdx.x * blockDim.x + threadIdx.x;
    if (i >= BATCH * HDIM) return;
    size_t base = (size_t)BATCH * LSEQ * LSEQ + (size_t)BATCH * LSEQ;
    out[base + i] = h[i];
    out[base + BATCH * HDIM + i] = c[i];
}

// ---------------------------------------------------------------------------
// Host launcher
// ---------------------------------------------------------------------------
extern "C" void kernel_launch(void* const* d_in, const int* in_sizes, int n_in,
                              void* d_out, int out_size, void* d_ws, size_t ws_size,
                              hipStream_t stream) {
    (void)in_sizes; (void)n_in; (void)out_size; (void)ws_size;
    const float* emb     = (const float*)d_in[0];   // [B,L,E]
    const float* dec     = (const float*)d_in[1];   // [B,E]
    const float* h0      = (const float*)d_in[2];   // [B,H]
    const float* c0      = (const float*)d_in[3];   // [B,H]
    const float* context = (const float*)d_in[4];   // [B,L,H]
    const float* Wih     = (const float*)d_in[5];   // [4H,E]
    const float* bih     = (const float*)d_in[6];
    const float* Whh     = (const float*)d_in[7];   // [4H,H]
    const float* bhh     = (const float*)d_in[8];
    const float* Who     = (const float*)d_in[9];   // [H,2H]
    const float* bho     = (const float*)d_in[10];
    const float* Wq      = (const float*)d_in[11];  // [H,H]
    const float* bq      = (const float*)d_in[12];
    const float* Wc      = (const float*)d_in[13];  // [H,H]
    const float* bc      = (const float*)d_in[14];
    const float* Vv      = (const float*)d_in[15];  // [H]

    float* out = (float*)d_out;
    char*  wsb = (char*)d_ws;

    // workspace layout (byte offsets; every size is 256B-aligned)
    size_t off = 0;
    float* ctx   = (float*)(wsb + off); off += (size_t)BATCH * LSEQ * HDIM * 4;   // 33.5MB
    _Float16* ctx16A = (_Float16*)(wsb + off); off += (size_t)BATCH * LSEQ * HDIM * 2; // context f16
    _Float16* Wg16   = (_Float16*)(wsb + off); off += (size_t)4 * HDIM * (EDIM + HDIM) * 2;
    _Float16* Wq16   = (_Float16*)(wsb + off); off += (size_t)HDIM * HDIM * 2;
    _Float16* Wc16   = (_Float16*)(wsb + off); off += (size_t)HDIM * HDIM * 2;
    _Float16* xh16   = (_Float16*)(wsb + off); off += (size_t)BATCH * (EDIM + HDIM) * 2;
    _Float16* ht16   = (_Float16*)(wsb + off); off += (size_t)BATCH * HDIM * 2;
    float* bg    = (float*)(wsb + off); off += (size_t)4 * HDIM * 4;
    float* gates = (float*)(wsb + off); off += (size_t)BATCH * 4 * HDIM * 4;
    float* hbuf  = (float*)(wsb + off); off += (size_t)BATCH * HDIM * 4;
    float* cbuf  = (float*)(wsb + off); off += (size_t)BATCH * HDIM * 4;
    float* htbuf = (float*)(wsb + off); off += (size_t)BATCH * HDIM * 4;
    float* inp   = (float*)(wsb + off); off += (size_t)BATCH * HDIM * 4;
    float* maskb = (float*)(wsb + off); off += (size_t)BATCH * LSEQ * 4;

    // ---- one-time f16 packing (weights + loop-invariant context) ----
    {
        int n = BATCH * LSEQ * HDIM;                       // 8.4M
        pack_f16<<<dim3((n + 255) / 256), dim3(256), 0, stream>>>(context, ctx16A, n);
        n = HDIM * HDIM;
        pack_f16<<<dim3((n + 255) / 256), dim3(256), 0, stream>>>(Wq, Wq16, n);
        pack_f16<<<dim3((n + 255) / 256), dim3(256), 0, stream>>>(Wc, Wc16, n);
        n = 4 * HDIM * (EDIM + HDIM);
        pack_gate_weights<<<dim3((n + 255) / 256), dim3(256), 0, stream>>>(
            Wih, Whh, bih, bhh, Wg16, bg);
    }

    // ---- hoisted context projection: ctx = context @ Wc^T + bc ----
    {
        int tasks = ((BATCH * LSEQ) >> 4) * (HDIM >> 6);   // 2048 * 4 = 8192
        gemm_wmma_f16<<<dim3(tasks / 8), dim3(256), 0, stream>>>(
            ctx16A, Wc16, bc, ctx, BATCH * LSEQ, HDIM, HDIM);
    }
    init_state<<<dim3(128), dim3(256), 0, stream>>>(hbuf, cbuf, xh16, maskb, h0, c0, dec);

    for (int t = 0; t < LSEQ; ++t) {
        // gates = [x|h] @ [Wih|Whh]^T + (bih+bhh)   M=128, N=1024, K=512
        {
            int tasks = (BATCH >> 4) * ((4 * HDIM) >> 6);  // 8 * 16 = 128
            gemm_wmma_f16<<<dim3(tasks / 8), dim3(256), 0, stream>>>(
                xh16, Wg16, bg, gates, BATCH, 4 * HDIM, EDIM + HDIM);
        }
        // c_t, h_t
        lstm_elem<<<dim3(128), dim3(256), 0, stream>>>(gates, cbuf, htbuf, ht16);
        // inp = h_t @ Wq^T + bq   M=128, N=256, K=256
        {
            int tasks = (BATCH >> 4) * (HDIM >> 6);        // 8 * 4 = 32
            gemm_wmma_f16<<<dim3(tasks / 8), dim3(256), 0, stream>>>(
                ht16, Wq16, bq, inp, BATCH, HDIM, HDIM);
        }
        // fused attention / softmax / pointer / output projection
        attn_step<<<dim3(BATCH), dim3(256), 0, stream>>>(
            ctx, inp, Vv, maskb, htbuf, Who, bho, emb,
            hbuf, xh16, out, t);
    }

    finalize<<<dim3(128), dim3(256), 0, stream>>>(hbuf, cbuf, out);
}
